// GCN_34187939676601
// MI455X (gfx1250) — compile-verified, matrix operand order checked by reading, not run on hardware
//
#include <hip/hip_runtime.h>

// ---- CDNA5 (gfx1250) wave32 WMMA types -----------------------------------
typedef __attribute__((ext_vector_type(16))) __bf16 v16bf;
typedef __attribute__((ext_vector_type(8)))  __bf16 v8bf;
typedef __attribute__((ext_vector_type(4)))  __bf16 v4bf;
typedef __attribute__((ext_vector_type(8)))  float  v8f;

#define BM 64          // block tile rows (4 waves x 16-row strips)
#define BK 32          // K step == WMMA bf16 K
#define THREADS 128    // 4 wave32 waves
#define LDSTR 40       // LDS row stride (bf16): 16B-aligned rows, 20-bank
                       // lane stride -> conflict-free 16B fragment loads

// C[M,F] = act( A[M,K] * B ), block tile BM x (NT*16).
//   TRANS_B=0: B is [K,F] row-major        (A @ H)
//   TRANS_B=1: B is [F,K] row-major, A@B^T (H @ W^T)
// fp32 in HBM, bf16 in LDS (converted while staging), f32 WMMA accumulate.
// Double-buffered LDS, register-staged global loads, 1 barrier / K-step,
// branchless OOB handling (address clamp + v_cndmask zeroing).
template <int NT, bool TRANS_B, bool RELU>
__global__ __launch_bounds__(THREADS) void gemm_wmma_bf16(
    const float* __restrict__ A, const float* __restrict__ B,
    float* __restrict__ C, int M, int K, int F) {
  constexpr int BN = NT * 16;
  __shared__ __bf16 As[2][BM * LDSTR];
  __shared__ __bf16 Bs[2][BN * LDSTR];

  const int tid  = threadIdx.x;
  const int lane = tid & 31;
  const int wave = tid >> 5;        // 0..3 -> 16-row strip
  const int half = lane >> 4;       // 0/1
  const int mn   = lane & 15;       // row (A) / col (B,C) within 16

  const int rowBase = blockIdx.x * BM;
  const int colBase = blockIdx.y * BN;

  v8f acc[NT];
#pragma unroll
  for (int t = 0; t < NT; ++t)
#pragma unroll
    for (int i = 0; i < 8; ++i) acc[t][i] = 0.0f;

  float4 aReg[4];    // BM*BK / (THREADS*4)
  float4 bReg[NT];   // BN*BK / (THREADS*4)

  // ---- global -> registers (branchless: clamp address, cndmask zero) ----
  auto loadTiles = [&](int k0) {
#pragma unroll
    for (int c = 0; c < 4; ++c) {
      int linear = c * (THREADS * 4) + tid * 4;
      int r  = linear >> 5;             // /BK
      int kc = linear & (BK - 1);
      int grow = rowBase + r;  grow = grow < M ? grow : M - 1;
      int gk = k0 + kc;
      bool ok = gk < K;                 // K%4==0, gk%4==0 -> all-or-nothing
      int gkl = ok ? gk : 0;
      float4 v = *(const float4*)(A + (size_t)grow * K + gkl);
      v.x = ok ? v.x : 0.f;  v.y = ok ? v.y : 0.f;
      v.z = ok ? v.z : 0.f;  v.w = ok ? v.w : 0.f;
      aReg[c] = v;
    }
    if (TRANS_B) {
#pragma unroll
      for (int c = 0; c < NT; ++c) {
        int linear = c * (THREADS * 4) + tid * 4;
        int r  = linear >> 5;
        int kc = linear & (BK - 1);
        int gf = colBase + r;  gf = gf < F ? gf : F - 1;
        int gk = k0 + kc;
        bool ok = gk < K;
        int gkl = ok ? gk : 0;
        float4 v = *(const float4*)(B + (size_t)gf * K + gkl);
        v.x = ok ? v.x : 0.f;  v.y = ok ? v.y : 0.f;
        v.z = ok ? v.z : 0.f;  v.w = ok ? v.w : 0.f;
        bReg[c] = v;
      }
    } else {
#pragma unroll
      for (int c = 0; c < NT; ++c) {
        int linear = c * (THREADS * 4) + tid * 4;
        int kr = linear / BN;
        int fc = linear & (BN - 1);
        int gk = k0 + kr;
        bool ok = gk < K;
        int gkr = ok ? gk : 0;
        float4 v = *(const float4*)(B + (size_t)gkr * F + colBase + fc);
        v.x = ok ? v.x : 0.f;  v.y = ok ? v.y : 0.f;
        v.z = ok ? v.z : 0.f;  v.w = ok ? v.w : 0.f;
        bReg[c] = v;
      }
    }
  };

  // ---- registers -> LDS (fp32 -> bf16) ----------------------------------
  auto storeTiles = [&](int buf) {
#pragma unroll
    for (int c = 0; c < 4; ++c) {
      int linear = c * (THREADS * 4) + tid * 4;
      int r  = linear >> 5;
      int kc = linear & (BK - 1);
      v4bf bv;
      bv[0] = (__bf16)aReg[c].x; bv[1] = (__bf16)aReg[c].y;
      bv[2] = (__bf16)aReg[c].z; bv[3] = (__bf16)aReg[c].w;
      *(v4bf*)(&As[buf][r * LDSTR + kc]) = bv;
    }
    if (TRANS_B) {
#pragma unroll
      for (int c = 0; c < NT; ++c) {
        int linear = c * (THREADS * 4) + tid * 4;
        int r  = linear >> 5;
        int kc = linear & (BK - 1);
        v4bf bv;
        bv[0] = (__bf16)bReg[c].x; bv[1] = (__bf16)bReg[c].y;
        bv[2] = (__bf16)bReg[c].z; bv[3] = (__bf16)bReg[c].w;
        *(v4bf*)(&Bs[buf][r * LDSTR + kc]) = bv;
      }
    } else {
#pragma unroll
      for (int c = 0; c < NT; ++c) {
        int linear = c * (THREADS * 4) + tid * 4;
        int kr = linear / BN;
        int fc = linear & (BN - 1);
        Bs[buf][(fc + 0) * LDSTR + kr] = (__bf16)bReg[c].x;
        Bs[buf][(fc + 1) * LDSTR + kr] = (__bf16)bReg[c].y;
        Bs[buf][(fc + 2) * LDSTR + kr] = (__bf16)bReg[c].z;
        Bs[buf][(fc + 3) * LDSTR + kr] = (__bf16)bReg[c].w;
      }
    }
  };

  // ---- LDS -> WMMA, one-ahead B-fragment ping-pong ----------------------
  union BF { v16bf v; v8bf h[2]; };
  auto compute = [&](int buf) {
    BF af, b0, b1;
    const __bf16* arow = &As[buf][(wave * 16 + mn) * LDSTR];
    af.h[0] = *(const v8bf*)(arow + half * 8);       // K =  half*8 ..
    af.h[1] = *(const v8bf*)(arow + 16 + half * 8);  // K = 16+half*8 ..
    {
      const __bf16* brow = &Bs[buf][mn * LDSTR];
      b0.h[0] = *(const v8bf*)(brow + half * 8);
      b0.h[1] = *(const v8bf*)(brow + 16 + half * 8);
    }
#pragma unroll
    for (int t = 0; t < NT; ++t) {
      BF& use = (t & 1) ? b1 : b0;
      BF& pre = (t & 1) ? b0 : b1;
      if (t + 1 < NT) {
        const __bf16* brow = &Bs[buf][((t + 1) * 16 + mn) * LDSTR];
        pre.h[0] = *(const v8bf*)(brow + half * 8);
        pre.h[1] = *(const v8bf*)(brow + 16 + half * 8);
      }
      acc[t] = __builtin_amdgcn_wmma_f32_16x16x32_bf16(
          false, af.v, false, use.v, (short)0, acc[t], false, false);
    }
  };

  const int nk = (K + BK - 1) / BK;

  loadTiles(0);
  storeTiles(0);
  __syncthreads();

  for (int kt = 0; kt < nk; ++kt) {
    const int cur  = kt & 1;
    const bool more = (kt + 1) < nk;
    if (more) loadTiles((kt + 1) * BK);          // overlap HBM with WMMA
    if (kt + 2 < nk) {                           // warm L2 two tiles ahead
      int r = tid >> 1;
      int grw = rowBase + r;  grw = grw < M ? grw : M - 1;
      __builtin_prefetch(A + (size_t)grw * K + (kt + 2) * BK + ((tid & 1) << 4),
                         0, 1);
    }
    compute(cur);
    if (more) storeTiles(cur ^ 1);
    __syncthreads();
  }

  // ---- epilogue: C VGPR r -> M = r + 8*half, N = mn ---------------------
#pragma unroll
  for (int r = 0; r < 8; ++r) {
    int grow = rowBase + wave * 16 + r + 8 * half;
    if (grow < M) {
      float* crow = C + (size_t)grow * F + colBase + mn;
#pragma unroll
      for (int t = 0; t < NT; ++t) {
        float v = acc[t][r];
        if (RELU) v = v > 0.f ? v : 0.f;
        crow[t * 16] = v;
      }
    }
  }
}

static inline void launch_gemm(const float* A, const float* B, float* C,
                               int M, int K, int F, bool transB, bool relu,
                               hipStream_t stream) {
  dim3 block(THREADS);
  if (!transB) {                 // A @ H   (F = 256 or 128), BN = 128
    dim3 grid((M + BM - 1) / BM, F / 128);
    gemm_wmma_bf16<8, false, false><<<grid, block, 0, stream>>>(A, B, C, M, K, F);
  } else if (F >= 128) {         // relu(H @ W^T), BN = 128
    dim3 grid((M + BM - 1) / BM, F / 128);
    gemm_wmma_bf16<8, true, true><<<grid, block, 0, stream>>>(A, B, C, M, K, F);
  } else {                       // final H @ W^T (F = 64), BN = 64
    dim3 grid((M + BM - 1) / BM, F / 64);
    gemm_wmma_bf16<4, true, false><<<grid, block, 0, stream>>>(A, B, C, M, K, F);
  }
  (void)relu;
}

extern "C" void kernel_launch(void* const* d_in, const int* in_sizes, int n_in,
                              void* d_out, int out_size, void* d_ws, size_t ws_size,
                              hipStream_t stream) {
  const float* A_s = (const float*)d_in[0];
  const float* X_s = (const float*)d_in[1];
  const float* A_t = (const float*)d_in[2];
  const float* X_t = (const float*)d_in[3];
  const float* W1  = (const float*)d_in[4];
  const float* W2  = (const float*)d_in[5];
  const float* W3  = (const float*)d_in[6];

  const int DIN = 256, DHID = 128, DOUT = 64;
  const int N = in_sizes[1] / DIN;   // 10000

  float* buf0 = (float*)d_ws;                 // N x 256 fp32
  float* buf1 = buf0 + (size_t)N * DIN;       // N x 256 fp32
  float* Zs   = (float*)d_out;                // N x 64
  float* Zt   = Zs + (size_t)N * DOUT;        // N x 64

  const float* As_[2] = {A_s, A_t};
  const float* Xs_[2] = {X_s, X_t};
  float*       Zo_[2] = {Zs, Zt};

  for (int b = 0; b < 2; ++b) {
    const float* A = As_[b];
    launch_gemm(A, Xs_[b], buf0, N, N, DIN, false, false, stream);   // T0
    launch_gemm(buf0, W1, buf1, N, DIN, DIN, true, true, stream);    // H1
    launch_gemm(A, buf1, buf0, N, N, DIN, false, false, stream);     // T1
    launch_gemm(buf0, W2, buf1, N, DIN, DHID, true, true, stream);   // H2
    launch_gemm(A, buf1, buf0, N, N, DHID, false, false, stream);    // T2
    launch_gemm(buf0, W3, Zo_[b], N, DHID, DOUT, true, false, stream); // Z
  }
}
